// GCNPrediction_15109694947445
// MI455X (gfx1250) — matrix operand order repaired and checked by reading, not compiled, not run on hardware
//
#include <hip/hip_runtime.h>

// ---------------------------------------------------------------------------
// gfx1250 WMMA (wave32): v_wmma_f32_16x16x32_bf16
// ---------------------------------------------------------------------------
typedef __attribute__((ext_vector_type(16))) __bf16 v16bf;
typedef __attribute__((ext_vector_type(8)))  float  v8f;
typedef __attribute__((ext_vector_type(8)))  unsigned short u16x8;

union BFrag {
  u16x8 h[2];
  unsigned short u[16];
  v16bf v;
};
union AccF {
  v8f v;
  float f[8];
};

__device__ __forceinline__ unsigned short f2bf(float f) {
  unsigned int u = __float_as_uint(f);
  u += 0x7FFFu + ((u >> 16) & 1u);   // round-to-nearest-even
  return (unsigned short)(u >> 16);
}
__device__ __forceinline__ float bf2f(unsigned short u) {
  return __uint_as_float((unsigned int)u << 16);
}

__device__ __forceinline__ void atomicMaxF(float* addr, float v) {
  if (v >= 0.f) atomicMax((int*)addr, __float_as_int(v));
  else          atomicMin((unsigned int*)addr, __float_as_uint(v));
}

// copy 32 bytes/lane global(bf16) -> LDS via async DMA (no VGPR round trip)
__device__ __forceinline__ void async_copy32(unsigned ldsoff, const unsigned short* gsrc) {
  asm volatile(
      "global_load_async_to_lds_b128 %0, %1, off\n\t"
      "global_load_async_to_lds_b128 %0, %1, off offset:16"
      :: "v"(ldsoff), "v"(gsrc) : "memory");
}

// ---------------------------------------------------------------------------
// Generic WMMA GEMM: out[M,N] = op( A[M,K](bf16) * Bw[N,K]^T(bf16) + bias )
// 128 threads = 4 waves; 64x64 block tile; wave w -> rows w*16..+15.
// A staged to LDS with double-buffered global_load_async_to_lds_b128.
// OMODE: 0 = f32 store, 1 = bf16 store, 2 = f32 atomic-max (rows/ROWDIV fold)
// ---------------------------------------------------------------------------
template <int OMODE, bool RELU, int ROWDIV>
__global__ __launch_bounds__(128) void gemm_bf16_kernel(
    const unsigned short* __restrict__ A, const unsigned short* __restrict__ Bw,
    const float* __restrict__ bias, void* __restrict__ outv,
    int M, int N, int K, int Nstore, int ldout)
{
  __shared__ unsigned short sA[2][64 * 32];
  const int tid  = threadIdx.x;
  const int wave = tid >> 5;
  const int lane = tid & 31;
  const int lh   = lane & 15;
  const int kh   = lane >> 4;
  const int m0 = blockIdx.x * 64;
  const int n0 = blockIdx.y * 64;

  // per-lane async-copy addressing: row = tid/2, 16-elem chunk = tid&1
  const int arow  = tid >> 1;
  const int achnk = (tid & 1) * 16;
  const unsigned short* gA = A + (size_t)(m0 + arow) * K + achnk;

  AccF acc[4];
#pragma unroll
  for (int j = 0; j < 4; ++j)
#pragma unroll
    for (int e = 0; e < 8; ++e) acc[j].f[e] = 0.f;

  const int nk = K >> 5;
  async_copy32((unsigned)(size_t)&sA[0][arow * 32 + achnk], gA);

  for (int kt = 0; kt < nk; ++kt) {
    if (kt + 1 < nk) {
      async_copy32((unsigned)(size_t)&sA[(kt + 1) & 1][arow * 32 + achnk],
                   gA + (size_t)(kt + 1) * 32);
      asm volatile("s_wait_asynccnt 0x2" ::: "memory");
    } else {
      asm volatile("s_wait_asynccnt 0x0" ::: "memory");
    }
    __syncthreads();                       // tile kt visible to all waves

    BFrag af;
    {
      const unsigned short* p = &sA[kt & 1][(wave * 16 + lh) * 32 + kh * 8];
      af.h[0] = *(const u16x8*)(p);
      af.h[1] = *(const u16x8*)(p + 16);
    }
    const int k0 = kt * 32;
#pragma unroll
    for (int j = 0; j < 4; ++j) {
      BFrag bfr;
      const unsigned short* bp = Bw + (size_t)(n0 + j * 16 + lh) * K + k0 + kh * 8;
      bfr.h[0] = *(const u16x8*)(bp);
      bfr.h[1] = *(const u16x8*)(bp + 16);
      acc[j].v = __builtin_amdgcn_wmma_f32_16x16x32_bf16(
          false, af.v, false, bfr.v, (short)0, acc[j].v, false, false);
    }
    __syncthreads();                       // all waves done with buffer kt&1
  }

  const int mbase = m0 + wave * 16 + (kh ? 8 : 0);
#pragma unroll
  for (int j = 0; j < 4; ++j) {
    const int c = n0 + j * 16 + lh;
    if (c < Nstore) {
      const float bv = bias[c];
#pragma unroll
      for (int r = 0; r < 8; ++r) {
        float v = acc[j].f[r] + bv;
        if (RELU) v = fmaxf(v, 0.f);
        const int m = mbase + r;
        if (OMODE == 0) {
          ((float*)outv)[(size_t)m * ldout + c] = v;
        } else if (OMODE == 1) {
          ((unsigned short*)outv)[(size_t)m * ldout + c] = f2bf(v);
        } else {
          atomicMaxF(&((float*)outv)[(size_t)(m / ROWDIV) * ldout + c], v);
        }
      }
    }
  }
}

// ---------------------------------------------------------------------------
// s1 gather-GEMM: A row (b,t,k) = [ h[b, idx[b,t,k]], h[b,t] ]  (K=512, bf16)
// out[M,128](bf16) = relu(A * Bw^T + bias)
// ---------------------------------------------------------------------------
__global__ __launch_bounds__(128) void gemm_s1_kernel(
    const unsigned short* __restrict__ h, const int* __restrict__ idx,
    const unsigned short* __restrict__ Bw, const float* __restrict__ bias,
    unsigned short* __restrict__ out, int M, int N, int K, int T, int KNN)
{
  __shared__ unsigned short sA[2][64 * 32];
  __shared__ int srcN[64], srcC[64];
  const int tid  = threadIdx.x;
  const int wave = tid >> 5;
  const int lane = tid & 31;
  const int lh   = lane & 15;
  const int kh   = lane >> 4;
  const int m0 = blockIdx.x * 64;
  const int n0 = blockIdx.y * 64;

  if (tid < 64) {
    const int gr  = m0 + tid;
    const int b   = gr / (T * KNN);
    const int rem = gr - b * T * KNN;
    const int t   = rem / KNN;
    srcN[tid] = b * T + idx[gr];
    srcC[tid] = b * T + t;
  }
  __syncthreads();

  const int arow  = tid >> 1;
  const int achnk = (tid & 1) * 16;

  auto issue = [&](int kt) {
    const int k0 = kt * 32;
    const int srow = (k0 < 256) ? srcN[arow] : srcC[arow];
    const int scol = (k0 < 256) ? (k0 + achnk) : (k0 - 256 + achnk);
    async_copy32((unsigned)(size_t)&sA[kt & 1][arow * 32 + achnk],
                 h + (size_t)srow * 256 + scol);
  };

  AccF acc[4];
#pragma unroll
  for (int j = 0; j < 4; ++j)
#pragma unroll
    for (int e = 0; e < 8; ++e) acc[j].f[e] = 0.f;

  const int nk = K >> 5;
  issue(0);
  for (int kt = 0; kt < nk; ++kt) {
    if (kt + 1 < nk) {
      issue(kt + 1);
      asm volatile("s_wait_asynccnt 0x2" ::: "memory");
    } else {
      asm volatile("s_wait_asynccnt 0x0" ::: "memory");
    }
    __syncthreads();

    BFrag af;
    {
      const unsigned short* p = &sA[kt & 1][(wave * 16 + lh) * 32 + kh * 8];
      af.h[0] = *(const u16x8*)(p);
      af.h[1] = *(const u16x8*)(p + 16);
    }
    const int k0 = kt * 32;
#pragma unroll
    for (int j = 0; j < 4; ++j) {
      BFrag bfr;
      const unsigned short* bp = Bw + (size_t)(n0 + j * 16 + lh) * K + k0 + kh * 8;
      bfr.h[0] = *(const u16x8*)(bp);
      bfr.h[1] = *(const u16x8*)(bp + 16);
      acc[j].v = __builtin_amdgcn_wmma_f32_16x16x32_bf16(
          false, af.v, false, bfr.v, (short)0, acc[j].v, false, false);
    }
    __syncthreads();
  }

  const int mbase = m0 + wave * 16 + (kh ? 8 : 0);
#pragma unroll
  for (int j = 0; j < 4; ++j) {
    const int c = n0 + j * 16 + lh;
    const float bv = bias[c];
#pragma unroll
    for (int r = 0; r < 8; ++r) {
      const float v = fmaxf(acc[j].f[r] + bv, 0.f);
      out[(size_t)(mbase + r) * N + c] = f2bf(v);
    }
  }
}

// ---------------------------------------------------------------------------
// kNN (top-3) over H=256: dist = 2*<xi,xj> - |xi|^2 - |xj|^2, WMMA inner.
// One block per (b, 16-row tile); h is bf16.
// ---------------------------------------------------------------------------
__global__ __launch_bounds__(128) void knn_kernel(
    const unsigned short* __restrict__ h, const float* __restrict__ xx,
    int* __restrict__ idx, int T)
{
  __shared__ unsigned short sA[16 * 256];
  __shared__ float sxx[16];
  __shared__ float sD[4 * 256];
  __shared__ float sTV[4 * 16 * 3];
  __shared__ int   sTI[4 * 16 * 3];

  const int tid  = threadIdx.x;
  const int wave = tid >> 5;
  const int lane = tid & 31;
  const int lh   = lane & 15;
  const int kh   = lane >> 4;
  const int tilesPerB = T / 16;
  const int b     = blockIdx.x / tilesPerB;
  const int trow0 = (blockIdx.x % tilesPerB) * 16;

  if (tid < 16) sxx[tid] = xx[b * T + trow0 + tid];
  {
    const int row = tid >> 3;
    const int c0  = (tid & 7) * 32;            // 32 elems = 64B per lane
    const unsigned short* src = h + (size_t)(b * T + trow0 + row) * 256 + c0;
    const unsigned ldsoff = (unsigned)(size_t)&sA[row * 256 + c0];
    asm volatile(
        "global_load_async_to_lds_b128 %0, %1, off\n\t"
        "global_load_async_to_lds_b128 %0, %1, off offset:16\n\t"
        "global_load_async_to_lds_b128 %0, %1, off offset:32\n\t"
        "global_load_async_to_lds_b128 %0, %1, off offset:48"
        :: "v"(ldsoff), "v"(src) : "memory");
  }
  asm volatile("s_wait_asynccnt 0x0" ::: "memory");
  __syncthreads();

  BFrag af[8];
#pragma unroll
  for (int ks = 0; ks < 8; ++ks) {
    const unsigned short* p = &sA[lh * 256 + ks * 32 + kh * 8];
    af[ks].h[0] = *(const u16x8*)(p);
    af[ks].h[1] = *(const u16x8*)(p + 16);
  }

  float tv0 = -1e30f, tv1 = -1e30f, tv2 = -1e30f;
  int   ti0 = 0, ti1 = 0, ti2 = 0;

  for (int j = wave; j < tilesPerB; j += 4) {
    AccF acc;
#pragma unroll
    for (int e = 0; e < 8; ++e) acc.f[e] = 0.f;
#pragma unroll
    for (int ks = 0; ks < 8; ++ks) {
      BFrag bfr;
      const unsigned short* p = h + (size_t)(b * T + j * 16 + lh) * 256 + ks * 32 + kh * 8;
      bfr.h[0] = *(const u16x8*)(p);
      bfr.h[1] = *(const u16x8*)(p + 16);
      acc.v = __builtin_amdgcn_wmma_f32_16x16x32_bf16(
          false, af[ks].v, false, bfr.v, (short)0, acc.v, false, false);
    }
    const float xn = xx[b * T + j * 16 + lh];
#pragma unroll
    for (int r = 0; r < 8; ++r) {
      const int mloc = kh ? (r + 8) : r;
      sD[wave * 256 + mloc * 16 + lh] = 2.f * acc.f[r] - sxx[mloc] - xn;
    }
    asm volatile("s_wait_dscnt 0x0" ::: "memory");   // intra-wave LDS RAW
    if (lane < 16) {
#pragma unroll
      for (int c = 0; c < 16; ++c) {
        const float d = sD[wave * 256 + lane * 16 + c];
        const int n = j * 16 + c;
        if (d > tv2) {
          if (d > tv0)      { tv2 = tv1; ti2 = ti1; tv1 = tv0; ti1 = ti0; tv0 = d; ti0 = n; }
          else if (d > tv1) { tv2 = tv1; ti2 = ti1; tv1 = d; ti1 = n; }
          else              { tv2 = d; ti2 = n; }
        }
      }
    }
  }

  if (lane < 16) {
    const int base = (wave * 16 + lane) * 3;
    sTV[base + 0] = tv0; sTI[base + 0] = ti0;
    sTV[base + 1] = tv1; sTI[base + 1] = ti1;
    sTV[base + 2] = tv2; sTI[base + 2] = ti2;
  }
  __syncthreads();
  if (wave == 0 && lane < 16) {
    float v0 = -1e30f, v1 = -1e30f, v2 = -1e30f;
    int i0 = 0, i1 = 0, i2 = 0;
    for (int w = 0; w < 4; ++w)
      for (int s = 0; s < 3; ++s) {
        const float d = sTV[(w * 16 + lane) * 3 + s];
        const int   n = sTI[(w * 16 + lane) * 3 + s];
        if (d > v2) {
          if (d > v0)      { v2 = v1; i2 = i1; v1 = v0; i1 = i0; v0 = d; i0 = n; }
          else if (d > v1) { v2 = v1; i2 = i1; v1 = d; i1 = n; }
          else             { v2 = d; i2 = n; }
        }
      }
    const int orow = (b * T + trow0 + lane) * 3;
    idx[orow + 0] = i0; idx[orow + 1] = i1; idx[orow + 2] = i2;
  }
}

// ---------------------------------------------------------------------------
// helper kernels (tiny VALU work)
// ---------------------------------------------------------------------------
__global__ void xx_kernel(const unsigned short* __restrict__ h, float* __restrict__ xx, int n) {
  const int r = blockIdx.x * blockDim.x + threadIdx.x;
  if (r >= n) return;
  const unsigned short* p = h + (size_t)r * 256;
  float s = 0.f;
  for (int i = 0; i < 256; i += 8) {
    u16x8 u = *(const u16x8*)(p + i);
#pragma unroll
    for (int e = 0; e < 8; ++e) { const float v = bf2f(u[e]); s += v * v; }
  }
  xx[r] = s;
}

// grouped temporal conv1d (k=3, pad=1) + relu; bf16 in/out, f32 weights
template <int CPG, int OPG>
__global__ void gconv_kernel(const unsigned short* __restrict__ in, const float* __restrict__ w,
                             const float* __restrict__ bias, unsigned short* __restrict__ out,
                             int T, int Cin, int Cout, int BT)
{
  const int gidx = blockIdx.x * blockDim.x + threadIdx.x;
  if (gidx >= BT * Cout) return;
  const int oc  = gidx % Cout;
  const int row = gidx / Cout;
  const int t   = row % T;
  const int g   = oc / OPG;
  const float* wp = w + (size_t)oc * CPG * 3;
  float s = bias[oc];
  for (int dt = -1; dt <= 1; ++dt) {
    const int tt = t + dt;
    if (tt < 0 || tt >= T) continue;
    const unsigned short* ip = in + (size_t)(row + dt) * Cin + g * CPG;
#pragma unroll
    for (int ic = 0; ic < CPG; ++ic) s += bf2f(ip[ic]) * wp[ic * 3 + dt + 1];
  }
  out[gidx] = f2bf(fmaxf(s, 0.f));
}

// s2: grouped 1x1 (G=32, 4-in/4-out per group) + relu; bf16 in/out
__global__ void s2_kernel(const unsigned short* __restrict__ in, const float* __restrict__ w,
                          const float* __restrict__ bias, unsigned short* __restrict__ out, int total)
{
  const int gidx = blockIdx.x * blockDim.x + threadIdx.x;
  if (gidx >= total) return;
  const int oc  = gidx & 127;
  const int row = gidx >> 7;
  const int g   = oc >> 2;
  const unsigned short* ip = in + (size_t)row * 128 + g * 4;
  const float* wp = w + oc * 4;
  float s = bias[oc] + bf2f(ip[0]) * wp[0] + bf2f(ip[1]) * wp[1]
                     + bf2f(ip[2]) * wp[2] + bf2f(ip[3]) * wp[3];
  out[gidx] = f2bf(fmaxf(s, 0.f));
}

__global__ void cvt_kernel(const float* __restrict__ src, unsigned short* __restrict__ dst, int n) {
  const int i = blockIdx.x * blockDim.x + threadIdx.x;
  if (i < n) dst[i] = f2bf(src[i]);
}

__global__ void cvt_pad_kernel(const float* __restrict__ src, unsigned short* __restrict__ dst,
                               int rows_src, int rows_dst, int k) {
  const int i = blockIdx.x * blockDim.x + threadIdx.x;
  if (i >= rows_dst * k) return;
  const int r = i / k;
  dst[i] = (r < rows_src) ? f2bf(src[i]) : (unsigned short)0;
}

__global__ void fill_kernel(unsigned int* __restrict__ p, unsigned int val, int n) {
  const int i = blockIdx.x * blockDim.x + threadIdx.x;
  if (i < n) p[i] = val;
}

// hnext(bf16) = relu(tout + sout + hcur)
__global__ void combine_kernel(const float* __restrict__ tout, const unsigned short* __restrict__ hin,
                               const float* __restrict__ sout, unsigned short* __restrict__ hout, int n) {
  const int i = blockIdx.x * blockDim.x + threadIdx.x;
  if (i < n) hout[i] = f2bf(fmaxf(tout[i] + bf2f(hin[i]) + sout[i], 0.f));
}

// ---------------------------------------------------------------------------
extern "C" void kernel_launch(void* const* d_in, const int* in_sizes, int n_in,
                              void* d_out, int out_size, void* d_ws, size_t ws_size,
                              hipStream_t stream) {
  (void)in_sizes; (void)n_in; (void)out_size; (void)ws_size;

  const float* x      = (const float*)d_in[0];
  const float* fcin_w = (const float*)d_in[1];
  const float* fcin_b = (const float*)d_in[2];
  const float* conv_w = (const float*)d_in[3];
  const float* conv_b = (const float*)d_in[4];
  const float* t1_w = (const float*)d_in[5];
  const float* t1_b = (const float*)d_in[6];
  const float* t2_w = (const float*)d_in[7];
  const float* t2_b = (const float*)d_in[8];
  const float* t3_w = (const float*)d_in[9];
  const float* t3_b = (const float*)d_in[10];
  const float* s1_w = (const float*)d_in[11];
  const float* s1_b = (const float*)d_in[12];
  const float* s2_w = (const float*)d_in[13];
  const float* s2_b = (const float*)d_in[14];
  const float* s3_w = (const float*)d_in[15];
  const float* s3_b = (const float*)d_in[16];
  const float* fc_w = (const float*)d_in[17];
  const float* fc_b = (const float*)d_in[18];
  float* outp = (float*)d_out;

  const int Bb = 16, Tt = 1024, Hc = 256, Wd = 128, KNN = 3;
  const int BT = Bb * Tt;        // 16384
  const int MS = BT * KNN;       // 49152

  char* wsp = (char*)d_ws;
  auto alloc = [&](size_t bytes) -> void* {
    void* p = (void*)wsp;
    wsp += (bytes + 255) & ~(size_t)255;
    return p;
  };
  typedef unsigned short bf16;
  bf16* xbf   = (bf16*)alloc((size_t)BT * 768 * 2);
  bf16* hbfA  = (bf16*)alloc((size_t)BT * Hc * 2);
  bf16* hbfB  = (bf16*)alloc((size_t)BT * Hc * 2);
  bf16* tb1   = (bf16*)alloc((size_t)BT * Wd * 2);
  bf16* tb2   = (bf16*)alloc((size_t)BT * Wd * 2);
  bf16* s1o   = (bf16*)alloc((size_t)MS * Wd * 2);
  bf16* s2o   = (bf16*)alloc((size_t)MS * Wd * 2);
  float* toutb = (float*)alloc((size_t)BT * Hc * 4);
  float* soutb = (float*)alloc((size_t)BT * Hc * 4);
  float* xxb   = (float*)alloc((size_t)BT * 4);
  int*   idxb  = (int*)alloc((size_t)MS * 4);
  bf16* fcinw_bf = (bf16*)alloc((size_t)256 * 768 * 2);
  bf16* t1w_bf   = (bf16*)alloc((size_t)2 * 128 * 256 * 2);
  bf16* t3w_bf   = (bf16*)alloc((size_t)2 * 256 * 128 * 2);
  bf16* s1w_bf   = (bf16*)alloc((size_t)2 * 128 * 512 * 2);
  bf16* s3w_bf   = (bf16*)alloc((size_t)2 * 256 * 128 * 2);
  bf16* fcw_bf   = (bf16*)alloc((size_t)64 * 256 * 2);

  auto cdiv = [](int a, int b) { return (a + b - 1) / b; };

  // activations + weights -> bf16
  cvt_kernel<<<cdiv(BT * 768, 256), 256, 0, stream>>>(x, xbf, BT * 768);
  cvt_kernel<<<cdiv(256 * 768, 256), 256, 0, stream>>>(fcin_w, fcinw_bf, 256 * 768);
  cvt_kernel<<<cdiv(2 * 128 * 256, 256), 256, 0, stream>>>(t1_w, t1w_bf, 2 * 128 * 256);
  cvt_kernel<<<cdiv(2 * 256 * 128, 256), 256, 0, stream>>>(t3_w, t3w_bf, 2 * 256 * 128);
  cvt_kernel<<<cdiv(2 * 128 * 512, 256), 256, 0, stream>>>(s1_w, s1w_bf, 2 * 128 * 512);
  cvt_kernel<<<cdiv(2 * 256 * 128, 256), 256, 0, stream>>>(s3_w, s3w_bf, 2 * 256 * 128);
  cvt_pad_kernel<<<cdiv(64 * 256, 256), 256, 0, stream>>>(fc_w, fcw_bf, 50, 64, 256);

  const dim3 gblk(128);
  // fc_in + relu -> hbfA (bf16)
  gemm_bf16_kernel<1, true, 1><<<dim3(BT / 64, 4), gblk, 0, stream>>>(
      xbf, fcinw_bf, fcin_b, hbfA, BT, 256, 768, 256, 256);
  // backbone grouped conv (G=4, 64ch/group) + relu -> hbfB
  gconv_kernel<64, 64><<<cdiv(BT * Hc, 256), 256, 0, stream>>>(
      hbfA, conv_w, conv_b, hbfB, Tt, Hc, Hc, BT);

  bf16* hcur = hbfB;
  bf16* hnext = hbfA;
  for (int l = 0; l < 2; ++l) {
    // temporal branch
    gemm_bf16_kernel<1, true, 1><<<dim3(BT / 64, 2), gblk, 0, stream>>>(
        hcur, t1w_bf + (size_t)l * 128 * 256, t1_b + l * 128, tb1,
        BT, 128, 256, 128, 128);
    gconv_kernel<4, 4><<<cdiv(BT * Wd, 256), 256, 0, stream>>>(
        tb1, t2_w + (size_t)l * 128 * 4 * 3, t2_b + l * 128, tb2, Tt, Wd, Wd, BT);
    gemm_bf16_kernel<0, false, 1><<<dim3(BT / 64, 4), gblk, 0, stream>>>(
        tb2, t3w_bf + (size_t)l * 256 * 128, t3_b + l * 256, toutb,
        BT, 256, 128, 256, 256);
    // semantic branch
    xx_kernel<<<cdiv(BT, 256), 256, 0, stream>>>(hcur, xxb, BT);
    knn_kernel<<<Bb * (Tt / 16), 128, 0, stream>>>(hcur, xxb, idxb, Tt);
    gemm_s1_kernel<<<dim3(MS / 64, 2), gblk, 0, stream>>>(
        hcur, idxb, s1w_bf + (size_t)l * 128 * 512, s1_b + l * 128, s1o,
        MS, 128, 512, Tt, KNN);
    s2_kernel<<<cdiv(MS * Wd, 256), 256, 0, stream>>>(
        s1o, s2_w + (size_t)l * 128 * 4, s2_b + l * 128, s2o, MS * Wd);
    fill_kernel<<<cdiv(BT * Hc, 256), 256, 0, stream>>>((unsigned int*)soutb, 0xFF800000u, BT * Hc);
    // s3 with fused K-max via float atomic max (rows/3 fold)
    gemm_bf16_kernel<2, false, 3><<<dim3(MS / 64, 4), gblk, 0, stream>>>(
        s2o, s3w_bf + (size_t)l * 256 * 128, s3_b + l * 256, soutb,
        MS, 256, 128, 256, 256);
    combine_kernel<<<cdiv(BT * Hc, 256), 256, 0, stream>>>(toutb, hcur, soutb, hnext, BT * Hc);
    bf16* tswap = hcur; hcur = hnext; hnext = tswap;
  }
  // head: [BT,256] x [50(pad 64),256]^T -> out f32 [BT,50]
  gemm_bf16_kernel<0, false, 1><<<dim3(BT / 64, 1), gblk, 0, stream>>>(
      hcur, fcw_bf, fc_b, outp, BT, 64, 256, 50, 50);
}
